// FlaxHunyuanVideo15AttnBlock_78941498900868
// MI455X (gfx1250) — matrix operand order repaired and checked by reading, not compiled, not run on hardware
//
#include <hip/hip_runtime.h>
#include <hip/hip_bf16.h>
#include <math.h>

#define DEVI __device__ __forceinline__

typedef __bf16 bf16_t;
typedef __attribute__((ext_vector_type(8)))  __bf16 bf16x8;
typedef __attribute__((ext_vector_type(16))) __bf16 v16bf;
typedef __attribute__((ext_vector_type(8)))  float   v8f;

constexpr int   S      = 16384;   // tokens (16*32*32)
constexpr int   C      = 128;     // channels
constexpr int   NHW    = 1024;    // tokens per frame
constexpr float SCALE  = 0.08838834764831845f;  // 1/sqrt(128)
constexpr float LOG2E  = 1.4426950408889634f;
constexpr int   WSTRIDE = 136;    // padded LDS row stride (128 + 8) -> conflict-free b128 reads
constexpr int   KSTRIDE = 136;
constexpr int   VSTRIDE = 72;     // padded (64 + 8)
constexpr int   PSTRIDE = 72;

DEVI v8f wmma_bf16(v16bf a, v16bf b, v8f c) {
  return __builtin_amdgcn_wmma_f32_16x16x32_bf16(false, a, false, b, (short)0, c, false, false);
}

// 16 bf16 from two 16B-aligned groups of 8 (A- and B-frag: p1 = p0+16)
DEVI v16bf frag_cat(const bf16_t* p0, const bf16_t* p1) {
  bf16x8 lo = *(const bf16x8*)p0;
  bf16x8 hi = *(const bf16x8*)p1;
  v16bf r;
#pragma unroll
  for (int i = 0; i < 8; ++i) { r[i] = lo[i]; r[i + 8] = hi[i]; }
  return r;
}

// ---------------------------------------------------------------------------
// Kernel 1: RMSNorm (L2 over C) + Q/K/V projections.  Q,K bf16 row-major
// (Q pre-scaled by 1/sqrt(C)); V stored transposed Vt[ch][tok] bf16.
// ---------------------------------------------------------------------------
template <bool IS_Q, bool TRANSPOSED>
DEVI void proj_tile(const v16bf* afrag, const bf16_t* wlds, const float* bias,
                    bf16_t* dst, int token0, int col, int half) {
#pragma unroll
  for (int nt = 0; nt < 8; ++nt) {
    v8f acc = {};
#pragma unroll
    for (int kc = 0; kc < 4; ++kc) {
      const bf16_t* bp = wlds + (nt * 16 + col) * WSTRIDE + kc * 32 + half * 8;
      acc = wmma_bf16(afrag[kc], frag_cat(bp, bp + 16), acc);
    }
    const float bval = bias[nt * 16 + col];
    if (TRANSPOSED) {
      bf16x8 ov;
#pragma unroll
      for (int r = 0; r < 8; ++r) ov[r] = (bf16_t)(acc[r] + bval);
      const int ch = nt * 16 + col;
      const int tok = token0 + half * 8;
      *(bf16x8*)(dst + (size_t)ch * S + tok) = ov;
    } else {
#pragma unroll
      for (int r = 0; r < 8; ++r) {
        float v = acc[r] + bval;
        if (IS_Q) v *= SCALE;
        const int tok = token0 + r + half * 8;
        dst[(size_t)tok * C + nt * 16 + col] = (bf16_t)v;
      }
    }
  }
}

__global__ __launch_bounds__(256) void k_rmsnorm_qkv(
    const float* __restrict__ x, const float* __restrict__ gamma,
    const float* __restrict__ wq, const float* __restrict__ bq,
    const float* __restrict__ wk, const float* __restrict__ bk,
    const float* __restrict__ wv, const float* __restrict__ bv,
    bf16_t* __restrict__ Q, bf16_t* __restrict__ K, bf16_t* __restrict__ Vt) {
  __shared__ __align__(16) bf16_t ldsW[3 * C * WSTRIDE];

  const int tid = threadIdx.x;
  const float* wsrc[3] = {wq, wk, wv};
#pragma unroll
  for (int m = 0; m < 3; ++m) {
    const float* w = wsrc[m];
    for (int i = tid; i < C * C; i += 256) {
      const int in = i >> 7, out = i & 127;             // w[in][out]
      ldsW[m * C * WSTRIDE + out * WSTRIDE + in] = (bf16_t)w[i];  // transposed
    }
  }
  __syncthreads();

  const int lane = tid & 31, wave = tid >> 5;
  const int token0 = (blockIdx.x * 8 + wave) * 16;
  const int col = lane & 15, half = lane >> 4;

  // load x for exactly the channels this lane's A-frag slots need
  float xv[4][2][8];
  float ss = 0.f;
  const float* xr = x + (size_t)(token0 + col) * C;
#pragma unroll
  for (int kc = 0; kc < 4; ++kc)
#pragma unroll
    for (int h = 0; h < 2; ++h) {
      const int ch = kc * 32 + half * 8 + h * 16;
      const float4* p = (const float4*)(xr + ch);
      float4 a = p[0], b = p[1];
      xv[kc][h][0] = a.x; xv[kc][h][1] = a.y; xv[kc][h][2] = a.z; xv[kc][h][3] = a.w;
      xv[kc][h][4] = b.x; xv[kc][h][5] = b.y; xv[kc][h][6] = b.z; xv[kc][h][7] = b.w;
#pragma unroll
      for (int i = 0; i < 8; ++i) ss += xv[kc][h][i] * xv[kc][h][i];
    }
  ss += __shfl_xor(ss, 16, 32);                 // partner lane holds other 64 chans
  const float rs = sqrtf((float)C) / (sqrtf(ss) + 1e-8f);

  v16bf afrag[4];
#pragma unroll
  for (int kc = 0; kc < 4; ++kc)
#pragma unroll
    for (int h = 0; h < 2; ++h) {
      const int ch = kc * 32 + half * 8 + h * 16;
      const float4* gp = (const float4*)(gamma + ch);
      float4 g0 = gp[0], g1 = gp[1];
      const float gv[8] = {g0.x, g0.y, g0.z, g0.w, g1.x, g1.y, g1.z, g1.w};
#pragma unroll
      for (int i = 0; i < 8; ++i)
        afrag[kc][h * 8 + i] = (bf16_t)(xv[kc][h][i] * rs * gv[i]);
    }

  proj_tile<true,  false>(afrag, ldsW + 0 * C * WSTRIDE, bq, Q,  token0, col, half);
  proj_tile<false, false>(afrag, ldsW + 1 * C * WSTRIDE, bk, K,  token0, col, half);
  proj_tile<false, true >(afrag, ldsW + 2 * C * WSTRIDE, bv, Vt, token0, col, half);
}

// ---------------------------------------------------------------------------
// Kernel 2: frame-causal flash attention.  8 waves/block share 64-key K/V
// chunks staged in LDS; query block always lies inside one frame so the key
// range is a whole number of chunks (no per-element masking).
// ---------------------------------------------------------------------------
__global__ __launch_bounds__(256) void k_attn(
    const bf16_t* __restrict__ Q, const bf16_t* __restrict__ K,
    const bf16_t* __restrict__ Vt, bf16_t* __restrict__ O) {
  __shared__ __align__(16) bf16_t ldsK[64 * KSTRIDE];
  __shared__ __align__(16) bf16_t ldsV[128 * VSTRIDE];
  __shared__ __align__(16) bf16_t ldsP[8 * 16 * PSTRIDE];

  const int tid = threadIdx.x;
  const int lane = tid & 31, wave = tid >> 5;
  const int col = lane & 15, half = lane >> 4;

  const int q0 = blockIdx.x * 128;
  const int token0 = q0 + wave * 16;
  const int kend = ((q0 / NHW) + 1) * NHW;       // uniform across block

  // Q fragments (Q already pre-scaled by 1/sqrt(C))
  v16bf qfrag[4];
  const bf16_t* qr = Q + (size_t)(token0 + col) * C;
#pragma unroll
  for (int kc = 0; kc < 4; ++kc) {
    const bf16_t* p = qr + kc * 32 + half * 8;
    qfrag[kc] = frag_cat(p, p + 16);
  }

  v8f oacc[8];
  const v8f vzero = {};
#pragma unroll
  for (int nt = 0; nt < 8; ++nt) oacc[nt] = vzero;
  float mrow[8], lrow[8];
#pragma unroll
  for (int r = 0; r < 8; ++r) { mrow[r] = -INFINITY; lrow[r] = 0.f; }

  bf16_t* pw = ldsP + wave * 16 * PSTRIDE;

  for (int kb = 0; kb < kend; kb += 64) {
    // cooperative stage: K chunk (64x128) and Vt chunk (128x64), bf16
    {
      const int r = tid >> 2, q = tid & 3;
      const uint4* ks = (const uint4*)(K + (size_t)(kb + r) * C + q * 32);
      uint4* kd = (uint4*)(ldsK + r * KSTRIDE + q * 32);
#pragma unroll
      for (int i = 0; i < 4; ++i) kd[i] = ks[i];
      const int vr = tid >> 1, vh = tid & 1;
      const uint4* vs = (const uint4*)(Vt + (size_t)vr * S + kb + vh * 32);
      uint4* vd = (uint4*)(ldsV + vr * VSTRIDE + vh * 32);
#pragma unroll
      for (int i = 0; i < 4; ++i) vd[i] = vs[i];
    }
    __syncthreads();

    // S = Q . K^T : 4 D-tiles of 16 keys
    v8f sc[4];
#pragma unroll
    for (int j = 0; j < 4; ++j) {
      v8f s = {};
#pragma unroll
      for (int kc = 0; kc < 4; ++kc) {
        const bf16_t* bp = ldsK + (j * 16 + col) * KSTRIDE + kc * 32 + half * 8;
        s = wmma_bf16(qfrag[kc], frag_cat(bp, bp + 16), s);
      }
      sc[j] = s;
    }

    // online softmax (rows M = r + 8*half live across 16 lanes)
#pragma unroll
    for (int r = 0; r < 8; ++r) {
      float mx = sc[0][r];
#pragma unroll
      for (int j = 1; j < 4; ++j) mx = fmaxf(mx, sc[j][r]);
#pragma unroll
      for (int d = 1; d < 16; d <<= 1) mx = fmaxf(mx, __shfl_xor(mx, d, 32));
      const float mnew = fmaxf(mrow[r], mx);
      const float corr = exp2f((mrow[r] - mnew) * LOG2E);
      mrow[r] = mnew;
      float rsum = 0.f;
#pragma unroll
      for (int j = 0; j < 4; ++j) {
        const float p = exp2f((sc[j][r] - mnew) * LOG2E);
        sc[j][r] = p;
        rsum += p;
      }
#pragma unroll
      for (int d = 1; d < 16; d <<= 1) rsum += __shfl_xor(rsum, d, 32);
      lrow[r] = lrow[r] * corr + rsum;
#pragma unroll
      for (int nt = 0; nt < 8; ++nt) oacc[nt][r] = oacc[nt][r] * corr;
    }

    // D-layout -> A-layout reshape of P via wave-private LDS slice
#pragma unroll
    for (int j = 0; j < 4; ++j)
#pragma unroll
      for (int r = 0; r < 8; ++r)
        pw[(r + half * 8) * PSTRIDE + j * 16 + col] = (bf16_t)sc[j][r];

    v16bf pfrag[2];
#pragma unroll
    for (int kc2 = 0; kc2 < 2; ++kc2) {
      const bf16_t* p = pw + col * PSTRIDE + kc2 * 32 + half * 8;
      pfrag[kc2] = frag_cat(p, p + 16);
    }

    // O += P . V
#pragma unroll
    for (int nt = 0; nt < 8; ++nt)
#pragma unroll
      for (int kc2 = 0; kc2 < 2; ++kc2) {
        const bf16_t* bp = ldsV + (nt * 16 + col) * VSTRIDE + kc2 * 32 + half * 8;
        oacc[nt] = wmma_bf16(pfrag[kc2], frag_cat(bp, bp + 16), oacc[nt]);
      }

    __syncthreads();  // all LDS reads done before next chunk overwrites
  }

  float inv[8];
#pragma unroll
  for (int r = 0; r < 8; ++r) inv[r] = 1.0f / lrow[r];
#pragma unroll
  for (int nt = 0; nt < 8; ++nt)
#pragma unroll
    for (int r = 0; r < 8; ++r) {
      const int tok = token0 + r + half * 8;
      O[(size_t)tok * C + nt * 16 + col] = (bf16_t)(oacc[nt][r] * inv[r]);
    }
}

// ---------------------------------------------------------------------------
// Kernel 3: out = O @ wo + bo + x  (fp32 output, residual add)
// ---------------------------------------------------------------------------
__global__ __launch_bounds__(256) void k_oproj(
    const bf16_t* __restrict__ O, const float* __restrict__ wo,
    const float* __restrict__ bo, const float* __restrict__ x,
    float* __restrict__ out) {
  __shared__ __align__(16) bf16_t ldsW[C * WSTRIDE];
  const int tid = threadIdx.x;
  for (int i = tid; i < C * C; i += 256) {
    const int in = i >> 7, o = i & 127;
    ldsW[o * WSTRIDE + in] = (bf16_t)wo[i];
  }
  __syncthreads();

  const int lane = tid & 31, wave = tid >> 5;
  const int token0 = (blockIdx.x * 8 + wave) * 16;
  const int col = lane & 15, half = lane >> 4;

  v16bf afrag[4];
  const bf16_t* orow = O + (size_t)(token0 + col) * C;
#pragma unroll
  for (int kc = 0; kc < 4; ++kc) {
    const bf16_t* p = orow + kc * 32 + half * 8;
    afrag[kc] = frag_cat(p, p + 16);
  }

#pragma unroll
  for (int nt = 0; nt < 8; ++nt) {
    v8f acc = {};
#pragma unroll
    for (int kc = 0; kc < 4; ++kc) {
      const bf16_t* bp = ldsW + (nt * 16 + col) * WSTRIDE + kc * 32 + half * 8;
      acc = wmma_bf16(afrag[kc], frag_cat(bp, bp + 16), acc);
    }
    const int ch = nt * 16 + col;
    const float bval = bo[ch];
#pragma unroll
    for (int r = 0; r < 8; ++r) {
      const size_t idx = (size_t)(token0 + r + half * 8) * C + ch;
      out[idx] = acc[r] + bval + x[idx];
    }
  }
}

// ---------------------------------------------------------------------------
extern "C" void kernel_launch(void* const* d_in, const int* in_sizes, int n_in,
                              void* d_out, int out_size, void* d_ws, size_t ws_size,
                              hipStream_t stream) {
  (void)in_sizes; (void)n_in; (void)out_size; (void)ws_size;
  const float* x  = (const float*)d_in[0];
  const float* g  = (const float*)d_in[1];
  const float* wq = (const float*)d_in[2];
  const float* bq = (const float*)d_in[3];
  const float* wk = (const float*)d_in[4];
  const float* bk = (const float*)d_in[5];
  const float* wv = (const float*)d_in[6];
  const float* bv = (const float*)d_in[7];
  const float* wo = (const float*)d_in[8];
  const float* bo = (const float*)d_in[9];
  float* out = (float*)d_out;

  bf16_t* Qw  = (bf16_t*)d_ws;              // [S][C] bf16, pre-scaled
  bf16_t* Kw  = Qw  + (size_t)S * C;        // [S][C] bf16
  bf16_t* Vtw = Kw  + (size_t)S * C;        // [C][S] bf16 (transposed)
  bf16_t* Ow  = Vtw + (size_t)S * C;        // [S][C] bf16 attention output

  k_rmsnorm_qkv<<<128, 256, 0, stream>>>(x, g, wq, bq, wk, bk, wv, bv, Qw, Kw, Vtw);
  k_attn<<<128, 256, 0, stream>>>(Qw, Kw, Vtw, Ow);
  k_oproj<<<128, 256, 0, stream>>>(Ow, wo, bo, x, out);
}
